// BilinearGlobalLinformerDebugCapsuleFC_52965536694927
// MI455X (gfx1250) — matrix operand order repaired and checked by reading, not compiled, not run on hardware
//
#include <hip/hip_runtime.h>
#include <hip/hip_bf16.h>

// ---------------------------------------------------------------------------
// BilinearGlobalLinformerCapsuleFC for MI455X (gfx1250, wave32, WMMA + TDM).
//
// Roofline: dominant Linformer GEMM = 17.2 GFLOP; current_pose traffic 268 MB
// -> 11.5 us at 23.3 TB/s; staying memory-bound needs ~1.5 PFLOPS -> f16 WMMA
// (v_wmma_f32_16x16x32_f16, f32 accumulate). E_proj is pre-converted to f16
// and transposed ONCE (prep kernel), then each workgroup loads its 64 KB
// E-tile into LDS via the Tensor Data Mover (one tensor_load_to_lds with
// row padding) and amortizes it over 4 batches -> E L2 traffic 268 MB total.
// Routing (logits -> softmax over 32 parents -> weighted sum) also runs on
// WMMA; per-b results accumulate via relaxed agent-scope f32 atomics.
// ---------------------------------------------------------------------------

typedef __attribute__((ext_vector_type(16))) _Float16 v16h;
typedef __attribute__((ext_vector_type(8)))  float    v8f;
typedef unsigned int u32x4 __attribute__((ext_vector_type(4)));
typedef int          i32x8 __attribute__((ext_vector_type(8)));
typedef int          i32x4 __attribute__((ext_vector_type(4)));

#define SCALE_  0.25f     // D^-0.5, D=16
#define EPS_    1e-5f
#define NB      4         // batches per workgroup (E-tile reuse)

// LDS strides (elements). Padded so 16-row fragment accesses are bank-
// conflict free (stride mod 64 dwords == 4 -> distinct bank groups per row).
#define SE_STR  264   // sE : [128][264] f16  (E[n] transposed: [h][t])
#define SA_STR  264   // sA : [16][264]  f16  (A tile: [d][t])
#define SKE_STR 136   // sKe: [16][136]  f16  (Khat row-major [d][h])
#define SKT_STR 40    // sKt: [128][40]  f16  (Khat transposed [h][d], K-pad)
#define SQ_STR  40    // sQ : [32][40]   f16  (q, K-padded 16->32 with zeros)
#define SLT_STR 36    // sLt: [128][36]  f32  (logits transposed [h][j])
#define SP_STR  136   // sP : [32][136]  f16  (softmax probs [j][h])

#define OFF_SE   0
#define SZ_SE    (128*SE_STR*2)            // 67584 B
#define OFF_SA   (OFF_SE + SZ_SE)
#define SZ_SA    (16*SA_STR*2)             // 8448 B
#define OFF_SW   (OFF_SA + SZ_SA)
#define SZ_SW    (256*4)                   // 1024 B
#define OFF_SKE  (OFF_SW + SZ_SW)
#define SZ_SKE   (16*SKE_STR*2)            // 4352 B
#define OFF_SKT  (OFF_SKE + SZ_SKE)
#define SZ_SKT   (128*SKT_STR*2)           // 10240 B
#define OFF_SQ   (OFF_SKT + SZ_SKT)
#define SZ_SQ    (32*SQ_STR*2)             // 2560 B
#define OFF_SL   (OFF_SQ + SZ_SQ)
#define SZ_SL    (128*SLT_STR*4)           // 18432 B
#define OFF_SP   (OFF_SL + SZ_SL)
#define SZ_SP    (32*SP_STR*2)             // 8704 B
#define LDS_TOTAL (OFF_SP + SZ_SP)         // 121344 B (2 WGs / 320KB WGP)

// ws layout (bytes)
#define WS_QH    0                         // f16 [256][32][32] = 512 KB
#define WS_CAND  (512*1024)                // f32 [256][32][16] = 512 KB
#define WS_ET    (1024*1024)               // f16 [64][128][256] = 4 MB

// --- WMMA f16 fragment loaders (ISA 7.12.2 layouts) ------------------------
// A 16x32 f16: lane L (hf=L/16, M=L%16): v0..3 = K[8*hf..+8),
//              v4..7 = K[16+8*hf..+8)  (dense-A table)
__device__ inline v16h load_frag_a(const _Float16* row, int kbase, int hf) {
  union { v16h v; uint4 q[2]; } u;
  u.q[0] = *(const uint4*)(row + kbase + 8*hf);
  u.q[1] = *(const uint4*)(row + kbase + 16 + 8*hf);
  return u.v;
}
// B 32x16 f16 from N-major Bt[N][K]: lane half hf holds K[8*hf..+8) then
// K[16+8*hf..+8) (same k order as load_frag_a).
__device__ inline v16h load_frag_b(const _Float16* row, int kbase, int hf) {
  union { v16h v; uint4 q[2]; } u;
  u.q[0] = *(const uint4*)(row + kbase + 8*hf);
  u.q[1] = *(const uint4*)(row + kbase + 16 + 8*hf);
  return u.v;
}

// --- prep 1: q[b,j] = w_next[j] @ next_pose[b,j] (f16, K-padded) + zero cand
__global__ __launch_bounds__(256)
void caps_prep(const float* __restrict__ NP, const float* __restrict__ Wn,
               _Float16* __restrict__ Qh, float* __restrict__ cand) {
  int g = blockIdx.x * 256 + threadIdx.x;        // (b,j)
  int j = g & 31;
  const float* P = NP + g * 16;
  const float* W = Wn + j * 16;
  float q[16];
#pragma unroll
  for (int r = 0; r < 4; ++r)
#pragma unroll
    for (int c = 0; c < 4; ++c) {
      float acc = 0.f;
#pragma unroll
      for (int k = 0; k < 4; ++k) acc += W[r*4+k] * P[k*4+c];
      q[r*4+c] = acc;
    }
  _Float16* dst = Qh + g * 32;
#pragma unroll
  for (int e = 0; e < 16; ++e) dst[e] = (_Float16)q[e];
#pragma unroll
  for (int e = 16; e < 32; ++e) dst[e] = (_Float16)0.f;
  float* cz = cand + g * 16;
#pragma unroll
  for (int e = 0; e < 16; ++e) cz[e] = 0.f;
}

// --- prep 2: E_proj f32 [n][t=256][h=128] -> Et f16 [n][h][t] (LDS-tiled) --
__global__ __launch_bounds__(256)
void caps_prep_e(const float* __restrict__ E, _Float16* __restrict__ Et) {
  __shared__ float tile[32][33];
  const int n   = blockIdx.x;
  const int tid = threadIdx.x;
  const float*    src = E  + (size_t)n * (256*128);
  _Float16*       dst = Et + (size_t)n * (128*256);
#pragma unroll 1
  for (int tt = 0; tt < 32; ++tt) {
    int t0 = (tt >> 2) * 32, h0 = (tt & 3) * 32;
    {
      int r = tid >> 3, c4 = (tid & 7) * 4;
      float4 v = *(const float4*)(src + (t0 + r) * 128 + h0 + c4);
      tile[r][c4 + 0] = v.x; tile[r][c4 + 1] = v.y;
      tile[r][c4 + 2] = v.z; tile[r][c4 + 3] = v.w;
    }
    __syncthreads();
    {
      int c = tid >> 3, r4 = (tid & 7) * 4;
      union { _Float16 h[4]; uint2 u; } o;
#pragma unroll
      for (int jj = 0; jj < 4; ++jj) o.h[jj] = (_Float16)tile[r4 + jj][c];
      *(uint2*)(dst + (h0 + c) * 256 + t0 + r4) = o.u;
    }
    __syncthreads();
  }
}

// --- main: one workgroup per (b-group, n); E tile reused over NB batches ---
__global__ __launch_bounds__(256)
void caps_main(const float* __restrict__ X,       // current_pose (flat)
               const float* __restrict__ Wc,      // w_current [64][16]
               const _Float16* __restrict__ Et,   // f16 [64][128][256]
               const _Float16* __restrict__ Qh,   // [256][32][32]
               float* __restrict__ cand) {        // [256][32][16] accumulator
  extern __shared__ char smem[];
  _Float16* sE  = (_Float16*)(smem + OFF_SE);
  _Float16* sA  = (_Float16*)(smem + OFF_SA);
  float*    sW  = (float*)   (smem + OFF_SW);
  _Float16* sKe = (_Float16*)(smem + OFF_SKE);
  _Float16* sKt = (_Float16*)(smem + OFF_SKT);
  _Float16* sQ  = (_Float16*)(smem + OFF_SQ);
  float*    sLt = (float*)   (smem + OFF_SL);
  _Float16* sP  = (_Float16*)(smem + OFF_SP);

  const int tid  = threadIdx.x;
  const int bg   = blockIdx.x >> 6;       // batch group (4 b's)
  const int n    = blockIdx.x & 63;
  const int wave = tid >> 5;
  const int lane = tid & 31;
  const int lm   = lane & 15;
  const int hf   = lane >> 4;

#if __has_builtin(__builtin_amdgcn_tensor_load_to_lds)
  // ---- Tensor Data Mover: Et[n] (f16 [128][256], 64 KB) -> sE with a 16 B
  // pad after every 512 B row (SE_STR = 264). D# per ISA 8.3-8.6.
  // 6-arg builtin: (u32x4 g0, i32x8 g1, i32x4 g2, i32x4 g3, i32x8 g4, cpol)
  if (wave == 0) {
    unsigned lds_addr = (unsigned)(uintptr_t)(void*)smem + OFF_SE;
    unsigned long long ga =
        (unsigned long long)(uintptr_t)(Et + (size_t)n * (128*256));
    u32x4 g0;
    g0.x = 1u;                                     // count=1, user mode
    g0.y = lds_addr;                               // lds_addr (bytes)
    g0.z = (unsigned)(ga & 0xFFFFFFFFu);           // global_addr[31:0]
    g0.w = (unsigned)((ga >> 32) & 0x01FFFFFFu)    // global_addr[56:32]
           | (2u << 30);                           // type = 2 ("image")
    i32x8 g1;
    g1[0] = (int)((1u << 16)       // data_size = 1 -> 2-byte elements
                | (1u << 20)       // pad_enable
                | (6u << 22)       // pad_interval: 128 dwords = 512 B
                | (3u << 25));     // pad_amount:   4 dwords  = 16 B
    g1[1] = (int)(256u << 16);     // tensor_dim0 = 256 (low 16)
    g1[2] = (int)(128u << 16);     // dim0 hi=0 | tensor_dim1 = 128
    g1[3] = (int)(256u << 16);     // dim1 hi=0 | tile_dim0 = 256
    g1[4] = (int)(128u);           // tile_dim1 = 128 | tile_dim2 = 0
    g1[5] = (int)(256u);           // tensor_dim0_stride = 256 elems
    g1[6] = (int)(32768u << 16);   // stride0 hi=0 | tensor_dim1_stride lo
    g1[7] = 0;                     // tensor_dim1_stride hi
    i32x4 gz4 = {0, 0, 0, 0};                      // groups 2/3: unused (2-D)
    i32x8 gz8 = {0, 0, 0, 0, 0, 0, 0, 0};          // trailing group: unused
    __builtin_amdgcn_tensor_load_to_lds(g0, g1, gz4, gz4, gz8, 0);
  }
#else
  // fallback: vectorized manual copy with pad insertion
  for (int i4 = tid; i4 < 4096; i4 += 256) {
    int row = i4 >> 5, col = i4 & 31;
    *(uint4*)(sE + row * SE_STR + col * 8) =
        *(const uint4*)(Et + (size_t)n * (128*256) + row * 256 + col * 8);
  }
#endif

  // w_current block: flat-reshape maps child-capsule index m=(n&3)*16 + t/16
  sW[tid] = Wc[(n & 3) * 256 + tid];
  // zero K=16..31 pad of Khat^T so the logits WMMA (K padded to 32) is exact
  if (tid < 128) {
    uint4 z = {0, 0, 0, 0};
    *(uint4*)(sKt + tid * SKT_STR + 16) = z;
    *(uint4*)(sKt + tid * SKT_STR + 24) = z;
  }

#if __has_builtin(__builtin_amdgcn_tensor_load_to_lds)
  if (wave == 0) __builtin_amdgcn_s_wait_tensorcnt(0);
#endif
  __syncthreads();

#pragma unroll 1
  for (int ib = 0; ib < NB; ++ib) {
    const int b = bg * NB + ib;

    // q tile: one uint4 per thread (threads 0..127), pad-stride insertion
    if (tid < 128) {
      int row = tid >> 2, col = (tid & 3) * 8;
      *(uint4*)(sQ + row * SQ_STR + col) =
          *(const uint4*)(Qh + b * (32*32) + row * 32 + col);
    }
    // fused step 1: A[d][t] = X(4x4 blocks) @ w_current; two b128 LDS stores
    {
      int d  = tid >> 4;
      int mm = tid & 15;
      const float* xrow =
          X + (size_t)b * 262144 + d * 16384 + n * 256 + mm * 16;
      float x[16];
#pragma unroll
      for (int i = 0; i < 16; ++i) x[i] = xrow[i];
      const float* w = sW + mm * 16;
      union { _Float16 h[16]; uint4 q[2]; } o;
#pragma unroll
      for (int r = 0; r < 4; ++r)
#pragma unroll
        for (int c = 0; c < 4; ++c) {
          float acc = 0.f;
#pragma unroll
          for (int k = 0; k < 4; ++k) acc += x[r*4+k] * w[k*4+c];
          o.h[r*4+c] = (_Float16)acc;
        }
      _Float16* dst = sA + d * SA_STR + mm * 16;
      *(uint4*)(dst + 0) = o.q[0];
      *(uint4*)(dst + 8) = o.q[1];
    }
    __syncthreads();

    // GEMM-1: Khat(16x128) = A(16x256) @ E(256x128); wave w owns h-tile w
    {
      v8f acc = {};
      const _Float16* arow = sA + lm * SA_STR;
      const _Float16* brow = sE + (16 * wave + lm) * SE_STR;
#pragma unroll
      for (int ks = 0; ks < 8; ++ks) {
        v16h a  = load_frag_a(arow, 32 * ks, hf);
        v16h bb = load_frag_b(brow, 32 * ks, hf);
        acc = __builtin_amdgcn_wmma_f32_16x16x32_f16(false, a, false, bb,
                                                     (short)0, acc, false, false);
      }
      int h = 16 * wave + lm;
      union { _Float16 h[8]; uint4 q; } kt;
#pragma unroll
      for (int v = 0; v < 8; ++v) {
        int d = v + 8 * hf;
        _Float16 kv = (_Float16)acc[v];
        sKe[d * SKE_STR + h] = kv;   // [d][h] : B-source for weighted sum
        kt.h[v] = kv;
      }
      // [h][d] rows: this lane owns d = 8*hf .. 8*hf+7 -> one b128 store
      *(uint4*)(sKt + h * SKT_STR + 8 * hf) = kt.q;
    }
    __syncthreads();

    // logits: L(32x128) = Q(32x16,pad32) @ Khat; stored transposed [h][j]
    {
      v16h bb = load_frag_b(sKt + (16 * wave + lm) * SKT_STR, 0, hf);
      int h = 16 * wave + lm;
#pragma unroll
      for (int jt = 0; jt < 2; ++jt) {
        v16h a  = load_frag_a(sQ + (jt * 16 + lm) * SQ_STR, 0, hf);
        v8f acc = {};
        acc = __builtin_amdgcn_wmma_f32_16x16x32_f16(false, a, false, bb,
                                                     (short)0, acc, false, false);
        // lane holds j = jt*16 + 8*hf .. +7 for fixed h -> two b128 stores
        union { v8f v; uint4 q[2]; } o; o.v = acc;
        float* dst = sLt + h * SLT_STR + jt * 16 + 8 * hf;
        *(uint4*)(dst + 0) = o.q[0];
        *(uint4*)(dst + 4) = o.q[1];
      }
    }
    __syncthreads();

    // softmax over the 32 parent capsules, per column h (contiguous row read)
    if (tid < 128) {
      int h = tid;
      const float4* row4 = (const float4*)(sLt + h * SLT_STR);
      float vals[32], m = -1e30f;
#pragma unroll
      for (int c = 0; c < 8; ++c) {
        float4 v = row4[c];
        vals[c*4+0] = v.x * SCALE_; vals[c*4+1] = v.y * SCALE_;
        vals[c*4+2] = v.z * SCALE_; vals[c*4+3] = v.w * SCALE_;
      }
#pragma unroll
      for (int j = 0; j < 32; ++j) m = fmaxf(m, vals[j]);
      float s = 0.f;
#pragma unroll
      for (int j = 0; j < 32; ++j) { float e = __expf(vals[j] - m); vals[j] = e; s += e; }
      float inv = 1.f / s;
#pragma unroll
      for (int j = 0; j < 32; ++j) sP[j * SP_STR + h] = (_Float16)(vals[j] * inv);
    }
    __syncthreads();

    // cand_partial(32x16) = P(32x128) @ Khat^T; K split across wave pairs
    {
      int jt = wave & 1;
      int ks = wave >> 1;
      v16h a  = load_frag_a(sP + (jt * 16 + lm) * SP_STR, 32 * ks, hf);
      v16h bb = load_frag_b(sKe + lm * SKE_STR, 32 * ks, hf);
      v8f acc = {};
      acc = __builtin_amdgcn_wmma_f32_16x16x32_f16(false, a, false, bb,
                                                   (short)0, acc, false, false);
      float* dst = cand + (size_t)b * 512;
#pragma unroll
      for (int v = 0; v < 8; ++v) {
        int j = jt * 16 + v + 8 * hf;
        __hip_atomic_fetch_add(dst + j * 16 + lm, acc[v],
                               __ATOMIC_RELAXED, __HIP_MEMORY_SCOPE_AGENT);
      }
    }
    __syncthreads();
  }
}

// --- finalize: w_next transform + LayerNorm --------------------------------
__global__ __launch_bounds__(256)
void caps_final(const float* __restrict__ cand, const float* __restrict__ Wn,
                const float* __restrict__ lnw, const float* __restrict__ lnb,
                float* __restrict__ out) {
  int g = blockIdx.x * 256 + threadIdx.x;        // (b,j)
  int j = g & 31;
  const float* C = cand + g * 16;
  const float* W = Wn + j * 16;
  float y[16], mu = 0.f;
#pragma unroll
  for (int r = 0; r < 4; ++r)
#pragma unroll
    for (int c = 0; c < 4; ++c) {
      float acc = 0.f;
#pragma unroll
      for (int k = 0; k < 4; ++k) acc += W[r*4+k] * C[k*4+c];
      y[r*4+c] = acc; mu += acc;
    }
  mu *= (1.f / 16.f);
  float var = 0.f;
#pragma unroll
  for (int e = 0; e < 16; ++e) { float d0 = y[e] - mu; var += d0 * d0; }
  var *= (1.f / 16.f);
  float inv = rsqrtf(var + EPS_);
  float* O = out + g * 16;
#pragma unroll
  for (int e = 0; e < 16; ++e) O[e] = (y[e] - mu) * inv * lnw[e] + lnb[e];
}

extern "C" void kernel_launch(void* const* d_in, const int* in_sizes, int n_in,
                              void* d_out, int out_size, void* d_ws, size_t ws_size,
                              hipStream_t stream) {
  const float* current_pose = (const float*)d_in[0];
  const float* next_pose    = (const float*)d_in[1];
  const float* w_current    = (const float*)d_in[2];
  const float* w_next       = (const float*)d_in[3];
  const float* E_proj       = (const float*)d_in[4];
  const float* ln_weight    = (const float*)d_in[5];
  const float* ln_bias      = (const float*)d_in[6];
  // d_in[7] = num_iter (reference performs a single routing pass)

  char* ws = (char*)d_ws;
  _Float16* Qh   = (_Float16*)(ws + WS_QH);
  float*    cand = (float*)   (ws + WS_CAND);
  _Float16* Et   = (_Float16*)(ws + WS_ET);

  caps_prep  <<<32,   256, 0,         stream>>>(next_pose, w_next, Qh, cand);
  caps_prep_e<<<64,   256, 0,         stream>>>(E_proj, Et);
  caps_main  <<<4096, 256, LDS_TOTAL, stream>>>(current_pose, w_current, Et, Qh, cand);
  caps_final <<<32,   256, 0,         stream>>>(cand, w_next, ln_weight, ln_bias, (float*)d_out);
}
